// AttentionDecoderRNN_85298050498867
// MI455X (gfx1250) — matrix operand order repaired and checked by reading, not compiled, not run on hardware
//
#include <hip/hip_runtime.h>
#include <hip/hip_bf16.h>
#include <math.h>

// ---------------------------------------------------------------------------
// AttentionDecoderRNN single-step forward for MI455X (gfx1250, wave32, WMMA)
//   B=64, H=EMB=1024, S=512, VOCAB=50257, LAYERS=2
// GEMMs: M=64; fp32 weights [N,K] row-major streamed once from HBM, converted
// to bf16 in registers (native packed cvt), v_wmma_f32_16x16x32_bf16 with f32
// accumulate. Each wave owns a 64x16 output stripe: B fragment is loaded and
// converted ONCE per K-step and reused by 4 consecutive WMMAs (4 M-tiles).
// ---------------------------------------------------------------------------

typedef __attribute__((ext_vector_type(16))) __bf16 v16bf;
typedef __attribute__((ext_vector_type(16))) float  v16f;
typedef __attribute__((ext_vector_type(8)))  float  v8f;

#define BB   64
#define HH   1024
#define SS   512
#define VOC  50257
#define G3H  3072

static __device__ __forceinline__ __bf16 f2bf(float f) {
    return (__bf16)f;   // native v_cvt on gfx1250 (RNE)
}

// ---------------------------------------------------------------------------
// Embedding gather + bf16 conversion of last_hidden layers (A operands)
// ---------------------------------------------------------------------------
__global__ void pack_embed_kernel(const int* __restrict__ step,
                                  const float* __restrict__ emb,
                                  const float* __restrict__ lasth,
                                  __bf16* __restrict__ xbf,
                                  __bf16* __restrict__ ha0,
                                  __bf16* __restrict__ ha1) {
    int g = blockIdx.x * blockDim.x + threadIdx.x;      // 0..65535
    int b = g >> 10, k = g & 1023;
    int idx = step[b];
    xbf[g] = f2bf(emb[(size_t)idx * HH + k]);
    ha0[g] = f2bf(lasth[g]);
    ha1[g] = f2bf(lasth[BB * HH + g]);
}

// ---------------------------------------------------------------------------
// bf16-WMMA GEMM: C[64,N] = act(A[64,K](bf16) * W[N,K](f32)^T + bias)
// block = 256 (8 waves). Each wave: one N-tile of 16 columns, all 64 rows.
// grid.x = ceil(N/128).
//
// Fragment layouts used here (A/B k order is this code's contiguous choice, not the ISA table):
//  A 16x32 : lanes 0-15 M=lane, elems 0..15 = K+half*16.. (contiguous, matches B)
//  B 32x16 : lanes 0-15 N=lane, elems 0..15 = K(k0..k0+15); lanes16-31: K+16..31
//  C/D     : elem i -> M = i + 8*(lane>=16), N = lane&15
// ---------------------------------------------------------------------------
template <int ACT>   // 0 = none, 1 = tanh
__global__ __launch_bounds__(256)
void gemm_bf16_wmma(const __bf16* __restrict__ A, int lda,
                    const float* __restrict__ W,
                    const float* __restrict__ bias,
                    float* __restrict__ C, int ldc,
                    __bf16* __restrict__ Cbf, int ldbf, int bfoff,
                    int N, int K) {
    const int wave = threadIdx.x >> 5;
    const int lane = threadIdx.x & 31;
    const int half = lane >> 4;
    const int nl   = lane & 15;
    const int col  = (blockIdx.x * 8 + wave) * 16 + nl;
    const bool colok = (col < N);

    const __bf16* Abase = A + (size_t)nl * lda;     // row (mi*16 + nl) via offset
    const float*  Wrow  = W + (size_t)col * K;

    v8f acc[4];
#pragma unroll
    for (int m = 0; m < 4; ++m)
#pragma unroll
        for (int i = 0; i < 8; ++i) acc[m][i] = 0.0f;

    for (int k0 = 0; k0 < K; k0 += 32) {
        // ---- B fragment: 16 consecutive K fp32 at k0 + half*16, convert once
        v16f wf;
        if (colok) {
            const float4* wp = (const float4*)(Wrow + k0 + half * 16);
            ((float4*)&wf)[0] = wp[0];
            ((float4*)&wf)[1] = wp[1];
            ((float4*)&wf)[2] = wp[2];
            ((float4*)&wf)[3] = wp[3];
        } else {
#pragma unroll
            for (int i = 0; i < 16; ++i) wf[i] = 0.0f;
        }
        v16bf bfrag = __builtin_convertvector(wf, v16bf);

        // ---- 4 A fragments (one per 16-row tile) x 4 WMMAs against bfrag
#pragma unroll
        for (int m = 0; m < 4; ++m) {
            const __bf16* Arow = Abase + (size_t)(m * 16) * lda;
            v16bf a;
            ((uint4*)&a)[0] = *(const uint4*)(Arow + k0 + half * 16);
            ((uint4*)&a)[1] = *(const uint4*)(Arow + k0 + half * 16 + 8);
            acc[m] = __builtin_amdgcn_wmma_f32_16x16x32_bf16(
                         false, a, false, bfrag, (short)0, acc[m], false, false);
        }
    }

    if (colok) {
        float bv = bias ? bias[col] : 0.0f;
#pragma unroll
        for (int m = 0; m < 4; ++m) {
#pragma unroll
            for (int i = 0; i < 8; ++i) {
                int row = m * 16 + half * 8 + i;
                float v = acc[m][i] + bv;
                if (ACT == 1) v = tanhf(v);
                C[(size_t)row * ldc + col] = v;
                if (Cbf) Cbf[(size_t)row * ldbf + bfoff + col] = f2bf(v);
            }
        }
    }
}

// ---------------------------------------------------------------------------
// GRU gate combine: h' = (1-z)*n + z*h_prev   (gates laid out [r|z|n] over 3H)
// ---------------------------------------------------------------------------
__global__ void gru_combine_kernel(const float* __restrict__ gi,
                                   const float* __restrict__ gh,
                                   const float* __restrict__ hprev,
                                   float* __restrict__ hout,
                                   __bf16* __restrict__ hbf, int bfld, int bfoff) {
    int g = blockIdx.x * blockDim.x + threadIdx.x;      // 0..65535
    int b = g >> 10, h = g & 1023;
    size_t base = (size_t)b * G3H;
    float ir = gi[base + h], iz = gi[base + HH + h], in_ = gi[base + 2 * HH + h];
    float hr = gh[base + h], hz = gh[base + HH + h], hn  = gh[base + 2 * HH + h];
    float r = 1.0f / (1.0f + expf(-(ir + hr)));
    float z = 1.0f / (1.0f + expf(-(iz + hz)));
    float n = tanhf(in_ + r * hn);
    float v = (1.0f - z) * n + z * hprev[g];
    hout[g] = v;
    hbf[(size_t)b * bfld + bfoff + h] = f2bf(v);
}

// ---------------------------------------------------------------------------
// scores[b,s] = dot(h1[b,:], enc[s,b,:])   one wave per (b,s)
// ---------------------------------------------------------------------------
__global__ void scores_kernel(const float* __restrict__ h1,
                              const float* __restrict__ enc,
                              float* __restrict__ scores) {
    int w    = (blockIdx.x * blockDim.x + threadIdx.x) >> 5;
    int lane = threadIdx.x & 31;
    int s = w & (SS - 1);
    int b = w >> 9;
    const float* e = enc + ((size_t)s * BB + b) * HH;
    const float* h = h1 + (size_t)b * HH;
    float sum = 0.0f;
#pragma unroll
    for (int t = 0; t < 8; ++t) {
        int off = t * 128 + lane * 4;
        float4 ev = *(const float4*)(e + off);
        float4 hv = *(const float4*)(h + off);
        sum += ev.x * hv.x + ev.y * hv.y + ev.z * hv.z + ev.w * hv.w;
    }
    for (int o = 16; o; o >>= 1) sum += __shfl_down(sum, o, 32);
    if (lane == 0) scores[(size_t)b * SS + s] = sum;
}

// softmax over S per batch row (in place), one block of 512 per b
__global__ void softmax_s_kernel(float* __restrict__ sc) {
    __shared__ float red[SS];
    int b = blockIdx.x, tid = threadIdx.x;
    float v = sc[(size_t)b * SS + tid];
    red[tid] = v; __syncthreads();
    for (int o = 256; o; o >>= 1) { if (tid < o) red[tid] = fmaxf(red[tid], red[tid + o]); __syncthreads(); }
    float m = red[0]; __syncthreads();
    float e = expf(v - m);
    red[tid] = e; __syncthreads();
    for (int o = 256; o; o >>= 1) { if (tid < o) red[tid] += red[tid + o]; __syncthreads(); }
    sc[(size_t)b * SS + tid] = e / red[0];
}

// context[b,h] = sum_s attn[b,s] * enc[s,b,h]; writes bf16 into Acat[:,1024+h]
__global__ void context_kernel(const float* __restrict__ attn,
                               const float* __restrict__ enc,
                               __bf16* __restrict__ acat) {
    __shared__ float sa[SS];
    int b = blockIdx.x >> 2, chunk = blockIdx.x & 3;
    int h = chunk * 256 + threadIdx.x;
    sa[threadIdx.x]       = attn[(size_t)b * SS + threadIdx.x];
    sa[threadIdx.x + 256] = attn[(size_t)b * SS + threadIdx.x + 256];
    __syncthreads();
    float acc = 0.0f;
#pragma unroll 4
    for (int s = 0; s < SS; ++s)
        acc += sa[s] * enc[((size_t)s * BB + b) * HH + h];
    acat[(size_t)b * 2048 + HH + h] = f2bf(acc);
}

// softmax over VOCAB per batch row (in place on d_out), one 1024-block per b
__global__ void softmax_vocab_kernel(float* __restrict__ out) {
    __shared__ float red[1024];
    int b = blockIdx.x, tid = threadIdx.x;
    float* row = out + (size_t)b * VOC;
    float m = -INFINITY;
    for (int i = tid; i < VOC; i += 1024) m = fmaxf(m, row[i]);
    red[tid] = m; __syncthreads();
    for (int o = 512; o; o >>= 1) { if (tid < o) red[tid] = fmaxf(red[tid], red[tid + o]); __syncthreads(); }
    m = red[0]; __syncthreads();
    float s = 0.0f;
    for (int i = tid; i < VOC; i += 1024) { float e = expf(row[i] - m); row[i] = e; s += e; }
    red[tid] = s; __syncthreads();
    for (int o = 512; o; o >>= 1) { if (tid < o) red[tid] += red[tid + o]; __syncthreads(); }
    float inv = 1.0f / red[0];
    for (int i = tid; i < VOC; i += 1024) row[i] *= inv;
}

// ---------------------------------------------------------------------------
extern "C" void kernel_launch(void* const* d_in, const int* in_sizes, int n_in,
                              void* d_out, int out_size, void* d_ws, size_t ws_size,
                              hipStream_t stream) {
    const int*   step   = (const int*)  d_in[0];
    const float* lasth  = (const float*)d_in[1];
    const float* enc    = (const float*)d_in[2];
    const float* emb    = (const float*)d_in[3];
    const float* w_ih0  = (const float*)d_in[4];
    const float* w_hh0  = (const float*)d_in[5];
    const float* b_ih0  = (const float*)d_in[6];
    const float* b_hh0  = (const float*)d_in[7];
    const float* w_ih1  = (const float*)d_in[8];
    const float* w_hh1  = (const float*)d_in[9];
    const float* b_ih1  = (const float*)d_in[10];
    const float* b_hh1  = (const float*)d_in[11];
    const float* ccw    = (const float*)d_in[12];
    const float* ccb    = (const float*)d_in[13];
    const float* out_w  = (const float*)d_in[14];
    const float* out_b  = (const float*)d_in[15];

    float* out_f   = (float*)d_out;                  // [64,50257] softmax output
    float* hid_out = out_f + (size_t)BB * VOC;       // [2,64,1024] new hidden
    float* hid0    = hid_out;
    float* hid1    = hid_out + BB * HH;

    // workspace layout (bytes, 256-aligned)
    char* ws = (char*)d_ws;
    __bf16* xbf   = (__bf16*)(ws + 0);               // [64,1024] bf16
    __bf16* ha0   = (__bf16*)(ws + (131072));        // [64,1024] bf16
    __bf16* ha1   = (__bf16*)(ws + (262144));        // [64,1024] bf16
    __bf16* h0bf  = (__bf16*)(ws + (393216));        // [64,1024] bf16
    __bf16* acat  = (__bf16*)(ws + (524288));        // [64,2048] bf16
    __bf16* cobf  = (__bf16*)(ws + (786432));        // [64,1024] bf16
    float*  gi    = (float*) (ws + (917504));        // [64,3072] f32
    float*  gh    = (float*) (ws + (917504 + 786432));      // [64,3072] f32
    float*  sc    = (float*) (ws + (917504 + 2 * 786432));  // [64,512]  f32
    float*  co    = (float*) (ws + (917504 + 2 * 786432 + 131072)); // [64,1024] f32

    dim3 blk256(256);

    // embedding gather + bf16 packs
    pack_embed_kernel<<<256, blk256, 0, stream>>>(step, emb, lasth, xbf, ha0, ha1);

    // ---- GRU layer 0 ----
    gemm_bf16_wmma<0><<<G3H / 128, blk256, 0, stream>>>(xbf, HH, w_ih0, b_ih0,
                                                        gi, G3H, nullptr, 0, 0, G3H, HH);
    gemm_bf16_wmma<0><<<G3H / 128, blk256, 0, stream>>>(ha0, HH, w_hh0, b_hh0,
                                                        gh, G3H, nullptr, 0, 0, G3H, HH);
    gru_combine_kernel<<<256, blk256, 0, stream>>>(gi, gh, lasth, hid0, h0bf, HH, 0);

    // ---- GRU layer 1 ----
    gemm_bf16_wmma<0><<<G3H / 128, blk256, 0, stream>>>(h0bf, HH, w_ih1, b_ih1,
                                                        gi, G3H, nullptr, 0, 0, G3H, HH);
    gemm_bf16_wmma<0><<<G3H / 128, blk256, 0, stream>>>(ha1, HH, w_hh1, b_hh1,
                                                        gh, G3H, nullptr, 0, 0, G3H, HH);
    // h1 -> hidden[1] (f32) and left half of Acat (bf16)
    gru_combine_kernel<<<256, blk256, 0, stream>>>(gi, gh, lasth + BB * HH, hid1,
                                                   acat, 2048, 0);

    // ---- Luong dot attention ----
    scores_kernel<<<(BB * SS) / 8, blk256, 0, stream>>>(hid1, enc, sc);
    softmax_s_kernel<<<BB, dim3(SS), 0, stream>>>(sc);
    context_kernel<<<BB * 4, blk256, 0, stream>>>(sc, enc, acat);   // right half of Acat

    // ---- concat layer: tanh([h1|ctx] @ concat_w.T + b) ----
    gemm_bf16_wmma<1><<<HH / 128, blk256, 0, stream>>>(acat, 2048, ccw, ccb,
                                                       co, HH, cobf, HH, 0, HH, 2048);

    // ---- output projection: logits straight into d_out ----
    gemm_bf16_wmma<0><<<(VOC + 127) / 128, blk256, 0, stream>>>(cobf, HH, out_w, out_b,
                                                                out_f, VOC, nullptr, 0, 0,
                                                                VOC, HH);
    // ---- softmax over vocab (in place) ----
    softmax_vocab_kernel<<<BB, dim3(1024), 0, stream>>>(out_f);
}
